// GatheringLoss_12489764896968
// MI455X (gfx1250) — matrix operand order repaired and checked by reading, not compiled, not run on hardware
//
#include <hip/hip_runtime.h>

typedef __attribute__((ext_vector_type(16))) __bf16 v16bf;
typedef __attribute__((ext_vector_type(8)))  float  v8f;

#define B_SZ 512
#define N_SZ 64
#define C_SZ 256
#define M_SZ 1024
#define TILE_M 32
#define ROWS_PER_BLOCK 128   // 8 waves * 16 rows
#define KCHUNKS 8            // 256 / 32

// ---------------------------------------------------------------------------
// Kernel 1: per-head GEMM (bf16 hi/lo split, 3x WMMA) + running argmax over M
// ---------------------------------------------------------------------------
__global__ __launch_bounds__(256)
void gl_score_argmax(const float* __restrict__ q,
                     const float* __restrict__ it,
                     int* __restrict__ idx_out) {
    __shared__ __align__(32) unsigned short s_hi[TILE_M][C_SZ];
    __shared__ __align__(32) unsigned short s_lo[TILE_M][C_SZ];

    const int n       = blockIdx.y;
    const int bBase   = blockIdx.x * ROWS_PER_BLOCK;
    const int tid     = threadIdx.x;
    const int lane    = tid & 31;
    const int wave    = tid >> 5;
    const int halfsel = lane >> 4;     // 0: lanes 0-15, 1: lanes 16-31
    const int l16     = lane & 15;

    // ---- Load this wave's 16x256 query strip into registers as split bf16,
    //      arranged per the 16-bit WMMA A-matrix layout (16x32 per K-chunk).
    v16bf a_hi[KCHUNKS], a_lo[KCHUNKS];
    {
        const int brow = bBase + wave * 16 + l16;
        const float* qrow = q + ((size_t)brow * N_SZ + n) * C_SZ;
        #pragma unroll
        for (int k = 0; k < KCHUNKS; ++k) {
            const int o0 = k * 32 + halfsel * 8;   // first 8-K run
            #pragma unroll
            for (int half = 0; half < 2; ++half) { // second run at +16
                const float4 f0 = *(const float4*)(qrow + o0 + half * 16);
                const float4 f1 = *(const float4*)(qrow + o0 + half * 16 + 4);
                float v[8] = {f0.x, f0.y, f0.z, f0.w, f1.x, f1.y, f1.z, f1.w};
                #pragma unroll
                for (int i = 0; i < 8; ++i) {
                    __bf16 h = (__bf16)v[i];
                    float  r = v[i] - (float)h;
                    a_hi[k][half * 8 + i] = h;
                    a_lo[k][half * 8 + i] = (__bf16)r;
                }
            }
        }
    }

    float best_s[8];
    int   best_m[8];
    #pragma unroll
    for (int r = 0; r < 8; ++r) { best_s[r] = -__builtin_inff(); best_m[r] = 0; }

    for (int m0 = 0; m0 < M_SZ; m0 += TILE_M) {
        __syncthreads();
        const float* tbase = it + ((size_t)n * M_SZ + m0) * C_SZ;

        // Prefetch next tile (one 128B line per thread -> 32 KB).
        if (m0 + TILE_M < M_SZ)
            __builtin_prefetch(tbase + TILE_M * C_SZ + tid * 32, 0, 1);

        // Cooperative load + f32 -> bf16 hi/lo split into LDS.
        #pragma unroll
        for (int j = 0; j < 8; ++j) {
            const int f  = j * 256 + tid;      // float4 index within tile
            const int ml = f >> 6;             // local m row (64 float4 per row)
            const int c0 = (f & 63) * 4;       // c offset
            const float4 d = *(const float4*)(tbase + ml * C_SZ + c0);
            float vv[4] = {d.x, d.y, d.z, d.w};
            unsigned short h[4], l[4];
            #pragma unroll
            for (int i = 0; i < 4; ++i) {
                union { __bf16 b; unsigned short u; } ch, cl;
                ch.b = (__bf16)vv[i];
                cl.b = (__bf16)(vv[i] - (float)ch.b);
                h[i] = ch.u; l[i] = cl.u;
            }
            *(unsigned int*)&s_hi[ml][c0]     = (unsigned int)h[0] | ((unsigned int)h[1] << 16);
            *(unsigned int*)&s_hi[ml][c0 + 2] = (unsigned int)h[2] | ((unsigned int)h[3] << 16);
            *(unsigned int*)&s_lo[ml][c0]     = (unsigned int)l[0] | ((unsigned int)l[1] << 16);
            *(unsigned int*)&s_lo[ml][c0 + 2] = (unsigned int)l[2] | ((unsigned int)l[3] << 16);
        }
        __syncthreads();

        // Two 16-column groups per tile; A registers fully reused.
        #pragma unroll
        for (int g = 0; g < 2; ++g) {
            const int col = g * 16 + l16;      // B-matrix column = local item row
            v8f acc = {};
            #pragma unroll
            for (int k = 0; k < KCHUNKS; ++k) {
                typedef __attribute__((ext_vector_type(8))) __bf16 v8bf_t;
                const v8bf_t bh0 = *(const v8bf_t*)&s_hi[col][k * 32 + halfsel * 8];
                const v8bf_t bh1 = *(const v8bf_t*)&s_hi[col][k * 32 + 16 + halfsel * 8];
                const v8bf_t bl0 = *(const v8bf_t*)&s_lo[col][k * 32 + halfsel * 8];
                const v8bf_t bl1 = *(const v8bf_t*)&s_lo[col][k * 32 + 16 + halfsel * 8];
                const v16bf bh = __builtin_shufflevector(bh0, bh1, 0,1,2,3,4,5,6,7,8,9,10,11,12,13,14,15);
                const v16bf bl = __builtin_shufflevector(bl0, bl1, 0,1,2,3,4,5,6,7,8,9,10,11,12,13,14,15);
                acc = __builtin_amdgcn_wmma_f32_16x16x32_bf16(false, a_hi[k], false, bh, (short)0, acc, false, false);
                acc = __builtin_amdgcn_wmma_f32_16x16x32_bf16(false, a_hi[k], false, bl, (short)0, acc, false, false);
                acc = __builtin_amdgcn_wmma_f32_16x16x32_bf16(false, a_lo[k], false, bh, (short)0, acc, false, false);
            }
            const int mcol = m0 + g * 16 + l16;
            #pragma unroll
            for (int r = 0; r < 8; ++r) {      // slot r: row = r + 8*halfsel
                const float s = acc[r];
                if (s > best_s[r]) { best_s[r] = s; best_m[r] = mcol; } // keeps first max
            }
        }
    }

    // Reduce the 16 column-candidates per row across the half-wave; tie -> lower m.
    #pragma unroll
    for (int r = 0; r < 8; ++r) {
        float s = best_s[r];
        int   mi = best_m[r];
        #pragma unroll
        for (int off = 1; off < 16; off <<= 1) {
            const float os = __shfl_xor(s, off, 32);
            const int   oi = __shfl_xor(mi, off, 32);
            if (os > s || (os == s && oi < mi)) { s = os; mi = oi; }
        }
        if (l16 == 0) {
            const int brow = bBase + wave * 16 + r + halfsel * 8;
            idx_out[brow * N_SZ + n] = mi;
        }
    }
}

// ---------------------------------------------------------------------------
// Kernel 2: gather selected item rows, per-block partial MSE sums (f32 exact)
// ---------------------------------------------------------------------------
__global__ __launch_bounds__(256)
void gl_gather_mse(const float* __restrict__ q,
                   const float* __restrict__ it,
                   const int* __restrict__ idx,
                   float* __restrict__ partial) {
    __shared__ float s_w[8];
    const int tid  = threadIdx.x;
    const int lane = tid & 31;
    const int wave = tid >> 5;
    const int p    = blockIdx.x * 8 + wave;    // pair id = b*N + n
    const int n    = p & (N_SZ - 1);
    const int mi   = idx[p];
    const float* qrow = q  + (size_t)p * C_SZ;
    const float* irow = it + ((size_t)n * M_SZ + mi) * C_SZ;

    float sum = 0.f;
    #pragma unroll
    for (int i = 0; i < 2; ++i) {
        const int c4 = lane + i * 32;
        const float4 qa = ((const float4*)qrow)[c4];
        const float4 ia = ((const float4*)irow)[c4];
        const float dx = qa.x - ia.x, dy = qa.y - ia.y;
        const float dz = qa.z - ia.z, dw = qa.w - ia.w;
        sum += dx * dx + dy * dy + dz * dz + dw * dw;
    }
    #pragma unroll
    for (int off = 1; off < 32; off <<= 1) sum += __shfl_xor(sum, off, 32);
    if (lane == 0) s_w[wave] = sum;
    __syncthreads();
    if (tid == 0) {
        float t = 0.f;
        #pragma unroll
        for (int w = 0; w < 8; ++w) t += s_w[w];
        partial[blockIdx.x] = t;
    }
}

// ---------------------------------------------------------------------------
// Kernel 3: deterministic final reduction -> mean
// ---------------------------------------------------------------------------
__global__ __launch_bounds__(256)
void gl_final_reduce(const float* __restrict__ partial, float* __restrict__ out) {
    __shared__ float s[256];
    const int tid = threadIdx.x;
    float t = 0.f;
    for (int i = tid; i < (B_SZ * N_SZ) / 8; i += 256) t += partial[i];
    s[tid] = t;
    __syncthreads();
    for (int w = 128; w > 0; w >>= 1) {
        if (tid < w) s[tid] += s[tid + w];
        __syncthreads();
    }
    if (tid == 0) out[0] = s[0] / (float)(B_SZ * N_SZ * C_SZ);
}

extern "C" void kernel_launch(void* const* d_in, const int* in_sizes, int n_in,
                              void* d_out, int out_size, void* d_ws, size_t ws_size,
                              hipStream_t stream) {
    (void)in_sizes; (void)n_in; (void)out_size; (void)ws_size;
    const float* q  = (const float*)d_in[0];
    const float* it = (const float*)d_in[1];
    int*   idx     = (int*)d_ws;                                     // 512*64*4 = 128 KB
    float* partial = (float*)((char*)d_ws + B_SZ * N_SZ * sizeof(int)); // 4096*4 = 16 KB

    dim3 g1(B_SZ / ROWS_PER_BLOCK, N_SZ);
    gl_score_argmax<<<g1, 256, 0, stream>>>(q, it, idx);
    gl_gather_mse<<<(B_SZ * N_SZ) / 8, 256, 0, stream>>>(q, it, idx, partial);
    gl_final_reduce<<<1, 256, 0, stream>>>(partial, (float*)d_out);
}
